// Mamba2Simple_46007689675001
// MI455X (gfx1250) — matrix operand order repaired and checked by reading, compiled, not run on hardware
//
#include <hip/hip_runtime.h>
#include <hip/hip_bf16.h>
#include <cstdint>

#define D_MODEL   1024
#define D_STATE   64
#define D_CONV    4
#define HEADDIM   128
#define D_INNER   2048
#define NHEADS    16
#define CONV_DIM  2176          // D_INNER + 2*D_STATE
#define D_IN_PROJ 4240          // 2*D_INNER + 2*D_STATE + NHEADS
#define NPROJ_PAD 4288          // 67 * 64 : padded rows for guard-free staging
#define BATCH     2
#define SEQLEN    4096
#define NTOK      (BATCH * SEQLEN)
#define EPS_RMS   1e-5f

typedef _Float16 half_t;
typedef __attribute__((ext_vector_type(16))) _Float16 v16h;
typedef __attribute__((ext_vector_type(8)))  float    v8f;

__device__ __forceinline__ float silu_f(float x) {
    return x / (1.0f + __expf(-x));
}
__device__ __forceinline__ float softplus_f(float x) {
    return (x > 20.0f) ? x : log1pf(__expf(x));
}

// ---------------------------------------------------------------------------
// fp32 -> f16 convert (zero-pads [n_src, n_total))
// ---------------------------------------------------------------------------
__global__ __launch_bounds__(256)
void f32_to_f16_kernel(const float* __restrict__ in, half_t* __restrict__ out,
                       long n_src, long n_total) {
    long i = (long)blockIdx.x * 256 + threadIdx.x;
    if (i < n_total)
        out[i] = (i < n_src) ? (half_t)in[i] : (half_t)0.0f;
}

// ---------------------------------------------------------------------------
// WMMA GEMM:  C[M x N] = X[M x K] * W[N x K]^T  (f16 operands, fp32 out)
// Block tile 128x64, 8 waves, each wave = 32x32 (2x2 v_wmma_f32_16x16x32_f16).
// DOUBLE-BUFFERED HBM->LDS staging via global_load_async_to_lds_b128:
// exactly 3 async instructions per wave per tile, async loads complete in
// order, so s_wait_asynccnt 0x3 = "previous tile resident" while the next
// tile's DMA is still in flight under the WMMAs.
// W must be padded so rows [N, gridDim.x*64) are readable (zero-filled).
// ---------------------------------------------------------------------------
#define BM  128
#define BN  64
#define BK  32
#define LDA 40   // 32 + 8 halves pad: 80B row stride, every b128 16B-aligned
#define LDB 40

__device__ __forceinline__
void stage_tile_async(const half_t* __restrict__ X, const half_t* __restrict__ W,
                      half_t* As, half_t* Bs, int tid,
                      int blockM, int blockN, int K, int k0) {
    // A tile: 128 rows x 32 halves, 4 x b128 per row, 2 passes x 256 lanes
#pragma unroll
    for (int c = 0; c < 2; ++c) {
        int lin = c * 256 + tid;        // 0..511
        int row = lin >> 2;
        int col = (lin & 3) * 8;        // in halves
        uint32_t lds = (uint32_t)(uintptr_t)&As[row * LDA + col];
        const half_t* g = X + (size_t)(blockM + row) * K + k0 + col;
        asm volatile("global_load_async_to_lds_b128 %0, %1, off"
                     :: "v"(lds), "v"(g) : "memory");
    }
    // B tile: 64 rows x 32 halves, 1 pass
    {
        int row = tid >> 2;
        int col = (tid & 3) * 8;
        uint32_t lds = (uint32_t)(uintptr_t)&Bs[row * LDB + col];
        const half_t* g = W + (size_t)(blockN + row) * K + k0 + col;
        asm volatile("global_load_async_to_lds_b128 %0, %1, off"
                     :: "v"(lds), "v"(g) : "memory");
    }
}

__global__ __launch_bounds__(256)
void gemm_f16_wmma(const half_t* __restrict__ X, const half_t* __restrict__ W,
                   float* __restrict__ C, int M, int N, int K) {
    __shared__ half_t As[2][BM * LDA];
    __shared__ half_t Bs[2][BN * LDB];

    const int tid    = threadIdx.x;
    const int lane   = tid & 31;
    const int wave   = tid >> 5;
    const int blockM = blockIdx.y * BM;
    const int blockN = blockIdx.x * BN;

    const int wm0 = (wave & 3) * 32;
    const int wn0 = (wave >> 2) * 32;
    const int mr  = lane & 15;
    const int sel = lane >> 4;

    v8f acc[2][2];
#pragma unroll
    for (int i = 0; i < 2; ++i)
#pragma unroll
        for (int j = 0; j < 2; ++j) acc[i][j] = (v8f){0,0,0,0,0,0,0,0};

    const int nk = K / BK;
    // prologue: stage tile 0 into buffer 0
    stage_tile_async(X, W, As[0], Bs[0], tid, blockM, blockN, K, 0);

    for (int ki = 0; ki < nk; ++ki) {
        const int buf = ki & 1;
        if (ki + 1 < nk) {
            // issue next tile into the other buffer, then wait only for the
            // current tile (3 newest async ops may remain outstanding)
            stage_tile_async(X, W, As[buf ^ 1], Bs[buf ^ 1], tid,
                             blockM, blockN, K, (ki + 1) * BK);
            asm volatile("s_wait_asynccnt 0x3" ::: "memory");
        } else {
            asm volatile("s_wait_asynccnt 0x0" ::: "memory");
        }
        __syncthreads();

        const half_t* A_ = As[buf];
        const half_t* B_ = Bs[buf];

        // ---- fragments per ISA 7.12.2 (16-bit A 16x32 / B 32x16)
        v16h afr[2], bfr[2];
#pragma unroll
        for (int i = 0; i < 2; ++i) {
            int m = wm0 + 16 * i + mr;
#pragma unroll
            for (int e = 0; e < 16; ++e) {
                int v  = e >> 1;
                int kl = (v >> 2) * 16 + sel * 8 + (v & 3) * 2 + (e & 1);
                afr[i][e] = A_[m * LDA + kl];
            }
        }
#pragma unroll
        for (int j = 0; j < 2; ++j) {
            int n = wn0 + 16 * j + mr;
#pragma unroll
            for (int e = 0; e < 16; ++e)
                bfr[j][e] = B_[n * LDB + (e >> 3) * 16 + sel * 8 + (e & 7)];
        }

#pragma unroll
        for (int i = 0; i < 2; ++i)
#pragma unroll
            for (int j = 0; j < 2; ++j)
                acc[i][j] = __builtin_amdgcn_wmma_f32_16x16x32_f16(
                    false, afr[i], false, bfr[j], (short)0, acc[i][j],
                    false, false);
        // all waves done reading this buffer before anyone refills it
        __syncthreads();
    }

#pragma unroll
    for (int i = 0; i < 2; ++i)
#pragma unroll
        for (int j = 0; j < 2; ++j)
#pragma unroll
            for (int r = 0; r < 8; ++r) {
                int gm = blockM + wm0 + 16 * i + sel * 8 + r;
                int gn = blockN + wn0 + 16 * j + mr;
                if (gn < N) C[(size_t)gm * N + gn] = acc[i][j][r];
            }
}

// ---------------------------------------------------------------------------
// Fused causal(fw)/anticausal(bw) depthwise conv(4)+SiLU, sliding 7-tap window.
// Thread owns one channel for a 64-long l-tile: each x element loaded once.
//   fw[l] = b + sum_k w[k] * x[l-3+k]   (window w[0..3])
//   bw[l] = b + sum_k w[k] * x[l+3-k]   (window w[6-k])
// ---------------------------------------------------------------------------
#define CONV_TILE_L 64

__global__ __launch_bounds__(256)
void conv_silu_kernel(const float* __restrict__ zx,
                      const float* __restrict__ wF, const float* __restrict__ bF,
                      const float* __restrict__ wB, const float* __restrict__ bB,
                      float* __restrict__ outF, float* __restrict__ outB) {
    int c = blockIdx.x * 256 + threadIdx.x;
    if (c >= CONV_DIM) return;
    const int tile = blockIdx.y;                     // NTOK / CONV_TILE_L tiles
    const int b    = tile / (SEQLEN / CONV_TILE_L);  // tiles never cross batch
    const int l0   = (tile % (SEQLEN / CONV_TILE_L)) * CONV_TILE_L;

    float wFv[D_CONV], wBv[D_CONV];
#pragma unroll
    for (int k = 0; k < D_CONV; ++k) {
        wFv[k] = wF[c * D_CONV + k];
        wBv[k] = wB[c * D_CONV + k];
    }
    const float bFv = bF[c], bBv = bB[c];

    float w[7];
#pragma unroll
    for (int j = 0; j < 7; ++j) {
        int l = l0 - 3 + j;
        w[j] = (l >= 0 && l < SEQLEN)
                 ? zx[((size_t)(b * SEQLEN + l)) * D_IN_PROJ + D_INNER + c]
                 : 0.0f;
    }

    for (int i = 0; i < CONV_TILE_L; ++i) {
        const int l = l0 + i;
        float aF = bFv, aB = bBv;
#pragma unroll
        for (int k = 0; k < D_CONV; ++k) {
            aF += wFv[k] * w[k];
            aB += wBv[k] * w[6 - k];
        }
        const size_t orow = (size_t)(b * SEQLEN + l) * CONV_DIM + c;
        outF[orow] = silu_f(aF);
        outB[orow] = silu_f(aB);
#pragma unroll
        for (int j = 0; j < 6; ++j) w[j] = w[j + 1];
        int ln = l + 4;
        w[6] = (ln < SEQLEN)
                 ? zx[((size_t)(b * SEQLEN + ln)) * D_IN_PROJ + D_INNER + c]
                 : 0.0f;
    }
}

// ---------------------------------------------------------------------------
// dt = softplus(zxbcdt[..., -NHEADS:] + dt_bias), both branches
// ---------------------------------------------------------------------------
__global__ __launch_bounds__(256)
void dt_kernel(const float* __restrict__ zx,
               const float* __restrict__ biasF, const float* __restrict__ biasB,
               float* __restrict__ dtF, float* __restrict__ dtB) {
    int idx = blockIdx.x * 256 + threadIdx.x;
    if (idx >= NTOK * NHEADS) return;
    int row = idx >> 4;
    int hd  = idx & (NHEADS - 1);
    float v = zx[(size_t)row * D_IN_PROJ + (D_IN_PROJ - NHEADS) + hd];
    dtF[idx] = softplus_f(v + biasF[hd]);
    dtB[idx] = softplus_f(v + biasB[hd]);
}

// ---------------------------------------------------------------------------
// SSD recurrence. One block per (head, batch); 128 lanes = HEADDIM channel p.
// h[64] per lane in VGPRs; B/C via LDS float4 broadcasts; 4 partial y sums.
// ---------------------------------------------------------------------------
__global__ __launch_bounds__(HEADDIM)
void ssd_scan_kernel(const float* __restrict__ conv_out,
                     const float* __restrict__ dt,
                     const float* __restrict__ A_log,
                     const float* __restrict__ Dp,
                     float* __restrict__ y, int dir) {
    const int hd = blockIdx.x;
    const int b  = blockIdx.y;
    const int p  = threadIdx.x;

    const float A  = -__expf(A_log[hd]);
    const float Dh = Dp[hd];

    float h[D_STATE];
#pragma unroll
    for (int n = 0; n < D_STATE; ++n) h[n] = 0.0f;

    __shared__ float sB[D_STATE];
    __shared__ float sC[D_STATE];
    __shared__ float sdt;

    for (int t = 0; t < SEQLEN; ++t) {
        const int l = (dir > 0) ? t : (SEQLEN - 1 - t);
        const size_t base = ((size_t)b * SEQLEN + l) * CONV_DIM;

        if (p < D_STATE)      sB[p]           = conv_out[base + D_INNER + p];
        else                  sC[p - D_STATE] = conv_out[base + D_INNER + D_STATE + (p - D_STATE)];
        if (p == 0)           sdt = dt[((size_t)b * SEQLEN + l) * NHEADS + hd];
        __syncthreads();

        const float x   = conv_out[base + hd * HEADDIM + p];
        const float dtv = sdt;
        const float dA  = __expf(dtv * A);
        const float dtx = dtv * x;

        const float4* sB4 = (const float4*)sB;
        const float4* sC4 = (const float4*)sC;
        float y0 = 0.f, y1 = 0.f, y2 = 0.f, y3 = 0.f;
#pragma unroll
        for (int q = 0; q < D_STATE / 4; ++q) {
            float4 bb = sB4[q];
            float4 cc = sC4[q];
            h[4*q+0] = h[4*q+0] * dA + dtx * bb.x;  y0 += h[4*q+0] * cc.x;
            h[4*q+1] = h[4*q+1] * dA + dtx * bb.y;  y1 += h[4*q+1] * cc.y;
            h[4*q+2] = h[4*q+2] * dA + dtx * bb.z;  y2 += h[4*q+2] * cc.z;
            h[4*q+3] = h[4*q+3] * dA + dtx * bb.w;  y3 += h[4*q+3] * cc.w;
        }
        y[((size_t)b * SEQLEN + l) * D_INNER + hd * HEADDIM + p] =
            (y0 + y1) + (y2 + y3) + Dh * x;
        __syncthreads();   // protect sB/sC for next step
    }
}

// ---------------------------------------------------------------------------
// Fused gating (SiLU(z)), per-branch RMSNorm, combine; emits f16 for GEMM2.
// ---------------------------------------------------------------------------
__global__ __launch_bounds__(256)
void gate_norm_kernel(const float* __restrict__ zx,
                      const float* __restrict__ yF, const float* __restrict__ yB,
                      const float* __restrict__ norm_w,
                      half_t* __restrict__ ycomb) {
    const int row = blockIdx.x;
    const int tid = threadIdx.x;

    float yzf[8], yzb[8];
    float sf = 0.0f, sb = 0.0f;
#pragma unroll
    for (int i = 0; i < 8; ++i) {
        int c = tid + i * 256;
        float g  = silu_f(zx[(size_t)row * D_IN_PROJ + c]);
        float vf = yF[(size_t)row * D_INNER + c] * g;
        float vb = yB[(size_t)row * D_INNER + c] * g;
        yzf[i] = vf;  yzb[i] = vb;
        sf += vf * vf;
        sb += vb * vb;
    }

    __shared__ float red[256];
    red[tid] = sf; __syncthreads();
    for (int s = 128; s > 0; s >>= 1) {
        if (tid < s) red[tid] += red[tid + s];
        __syncthreads();
    }
    const float rf = rsqrtf(red[0] / (float)D_INNER + EPS_RMS);
    __syncthreads();

    red[tid] = sb; __syncthreads();
    for (int s = 128; s > 0; s >>= 1) {
        if (tid < s) red[tid] += red[tid + s];
        __syncthreads();
    }
    const float rb = rsqrtf(red[0] / (float)D_INNER + EPS_RMS);

#pragma unroll
    for (int i = 0; i < 8; ++i) {
        int c = tid + i * 256;
        ycomb[(size_t)row * D_INNER + c] =
            (half_t)((yzf[i] * rf + yzb[i] * rb) * norm_w[c]);
    }
}

// ---------------------------------------------------------------------------
extern "C" void kernel_launch(void* const* d_in, const int* in_sizes, int n_in,
                              void* d_out, int out_size, void* d_ws, size_t ws_size,
                              hipStream_t stream) {
    (void)in_sizes; (void)n_in; (void)out_size; (void)ws_size;

    const float* u          = (const float*)d_in[0];
    const float* in_proj_w  = (const float*)d_in[1];
    const float* conv_w     = (const float*)d_in[2];
    const float* conv_b     = (const float*)d_in[3];
    const float* conv_w_bw  = (const float*)d_in[4];
    const float* conv_b_bw  = (const float*)d_in[5];
    const float* dt_bias    = (const float*)d_in[6];
    const float* dt_bias_bw = (const float*)d_in[7];
    const float* A_log      = (const float*)d_in[8];
    const float* A_b_log    = (const float*)d_in[9];
    const float* Dv         = (const float*)d_in[10];
    const float* D_b        = (const float*)d_in[11];
    const float* norm_w     = (const float*)d_in[12];
    const float* out_proj_w = (const float*)d_in[13];
    float* out = (float*)d_out;

    // ---- workspace layout
    const size_t ZX_N   = (size_t)NTOK * D_IN_PROJ;
    const size_t CONV_N = (size_t)NTOK * CONV_DIM;
    const size_t DT_N   = (size_t)NTOK * NHEADS;
    const size_t Y_N    = (size_t)NTOK * D_INNER;

    char* p = (char*)d_ws;
    float* zx    = (float*)p;  p += ZX_N   * sizeof(float);
    float* convF = (float*)p;  p += CONV_N * sizeof(float);
    float* convB = (float*)p;  p += CONV_N * sizeof(float);
    float* dtF   = (float*)p;  p += DT_N   * sizeof(float);
    float* dtB   = (float*)p;  p += DT_N   * sizeof(float);
    float* yF    = (float*)p;  p += Y_N    * sizeof(float);
    float* yB    = (float*)p;  p += Y_N    * sizeof(float);
    half_t* uh   = (half_t*)p; p += (size_t)NTOK * D_MODEL * sizeof(half_t);
    half_t* wh1  = (half_t*)p; p += (size_t)NPROJ_PAD * D_MODEL * sizeof(half_t);
    half_t* wh2  = (half_t*)p; p += (size_t)D_MODEL * D_INNER * sizeof(half_t);
    half_t* ycombh = (half_t*)convF;   // conv buffers dead after the scans

    // ---- 0) f16 operand copies (wh1 zero-padded to 4288 rows)
    {
        long n;
        n = (long)NTOK * D_MODEL;
        f32_to_f16_kernel<<<(n + 255) / 256, 256, 0, stream>>>(u, uh, n, n);
        long nsrc = (long)D_IN_PROJ * D_MODEL, ntot = (long)NPROJ_PAD * D_MODEL;
        f32_to_f16_kernel<<<(ntot + 255) / 256, 256, 0, stream>>>(in_proj_w, wh1, nsrc, ntot);
        n = (long)D_MODEL * D_INNER;
        f32_to_f16_kernel<<<(n + 255) / 256, 256, 0, stream>>>(out_proj_w, wh2, n, n);
    }

    // ---- 1) zxbcdt = u @ in_proj_w^T   (M=8192, N=4240, K=1024)
    {
        dim3 grid(NPROJ_PAD / BN, NTOK / BM);   // 67 x 64
        gemm_f16_wmma<<<grid, 256, 0, stream>>>(uh, wh1, zx,
                                                NTOK, D_IN_PROJ, D_MODEL);
    }
    // ---- 2) conv + SiLU, both branches (single-read sliding window)
    {
        dim3 grid((CONV_DIM + 255) / 256, NTOK / CONV_TILE_L);
        conv_silu_kernel<<<grid, 256, 0, stream>>>(zx, conv_w, conv_b,
                                                   conv_w_bw, conv_b_bw,
                                                   convF, convB);
    }
    // ---- 3) dt softplus
    dt_kernel<<<(NTOK * NHEADS + 255) / 256, 256, 0, stream>>>(
        zx, dt_bias, dt_bias_bw, dtF, dtB);

    // ---- 4) SSD scans (fw and reversed)
    {
        dim3 grid(NHEADS, BATCH);
        ssd_scan_kernel<<<grid, HEADDIM, 0, stream>>>(convF, dtF, A_log, Dv, yF, +1);
        ssd_scan_kernel<<<grid, HEADDIM, 0, stream>>>(convB, dtB, A_b_log, D_b, yB, -1);
    }
    // ---- 5) gate + RMSNorm + combine -> f16
    gate_norm_kernel<<<NTOK, 256, 0, stream>>>(zx, yF, yB, norm_w, ycombh);

    // ---- 6) out = y_comb @ out_proj_w^T  (M=8192, N=1024, K=2048)
    {
        dim3 grid(D_MODEL / BN, NTOK / BM);     // 16 x 64
        gemm_f16_wmma<<<grid, 256, 0, stream>>>(ycombh, wh2, out,
                                                NTOK, D_MODEL, D_INNER);
    }
}
